// StressLatentConditioned_21921513079372
// MI455X (gfx1250) — matrix operand order repaired and checked by reading, not compiled, not run on hardware
//
#include <hip/hip_runtime.h>
#include <math.h>

typedef __attribute__((ext_vector_type(16))) _Float16 v16h;
typedef __attribute__((ext_vector_type(8)))  _Float16 v8h;
typedef __attribute__((ext_vector_type(8)))  float    v8f;

#define NBLK_THREADS 128
#define TILES 4            // 16-row WMMA tiles per block (one per wave)
#define ROWS_PER_BLOCK 64  // batch elements per block
#define ASTRIDE 72         // f16 elems per activation row (padded vs 64)

// Raw hardware reciprocal (v_rcp_f32, ~1 ULP) -- far tighter than f16 storage.
__device__ __forceinline__ float fast_rcp(float x) {
    return __builtin_amdgcn_rcpf(x);
}

// Branch-free GELU (tanh form): v_exp_f32 + v_rcp_f32, no division expansion.
// Matches exact-erf GELU to ~1e-3, consistent with f16 activation storage.
__device__ __forceinline__ float gelu_fast(float x) {
    float x2 = x * x;
    float u  = x * (0.7978845608028654f + 0.03567740813636141f * x2);
    float e  = __expf(2.0f * u);                         // native exp
    float th = 1.0f - 2.0f * fast_rcp(e + 1.0f);         // tanh(u)
    return 0.5f * x * (1.0f + th);
}

__device__ __forceinline__ v16h cat8(v8h lo, v8h hi) {
    return __builtin_shufflevector(lo, hi, 0,1,2,3,4,5,6,7,8,9,10,11,12,13,14,15);
}

__global__ __launch_bounds__(NBLK_THREADS) void stress_mlp_kernel(
    const float* __restrict__ Fg, const float* __restrict__ Cg,
    const float* __restrict__ w1, const float* __restrict__ b1,
    const float* __restrict__ w2, const float* __restrict__ b2,
    const float* __restrict__ w3, const float* __restrict__ b3,
    const float* __restrict__ w4, const float* __restrict__ b4,
    const float* __restrict__ w5, const float* __restrict__ b5,
    const float* __restrict__ lat, const int* __restrict__ traj_ids,
    float* __restrict__ outg, int nElem)
{
    // Weights as f16: w1..w4 zero-padded to 64x64 (K-major), w5 padded to 64x16
    __shared__ __align__(16) _Float16 sW[4*64*64 + 64*16];
    __shared__ float sB[4*64 + 16];
    // Ping-pong activation tiles, one per wave: [buf][tile][row(M)][col(K/N)]
    __shared__ __align__(16) _Float16 sAct[2][TILES][16][ASTRIDE];

    const int t = threadIdx.x;

    // ---- stage weights into LDS (f16) ----
    for (int idx = t; idx < 64*64; idx += NBLK_THREADS) {
        int r = idx >> 6;
        sW[idx]          = (_Float16)((r < 57) ? w1[idx] : 0.0f);
        sW[4096  + idx]  = (_Float16)w2[idx];
        sW[8192  + idx]  = (_Float16)w3[idx];
        sW[12288 + idx]  = (_Float16)w4[idx];
    }
    for (int idx = t; idx < 64*16; idx += NBLK_THREADS) {
        int r = idx >> 4, c = idx & 15;
        sW[16384 + idx] = (_Float16)((c < 9) ? w5[r*9 + c] : 0.0f);
    }
    if (t < 64) { sB[t] = b1[t]; sB[64+t] = b2[t]; sB[128+t] = b3[t]; sB[192+t] = b4[t]; }
    if (t < 16) sB[256 + t] = (t < 9) ? b5[t] : 0.0f;

    // ---- per-element feature construction (threads 0..63) ----
    float Rm[9];   // polar rotation, kept for epilogue
    float Fm[9];   // F, kept for epilogue (F^T = Fm transposed on the fly)
    const int eIdx = blockIdx.x * ROWS_PER_BLOCK + t;

    if (t < ROWS_PER_BLOCK) {
        int e = (eIdx < nElem) ? eIdx : (nElem - 1);
        const float* fp = Fg + (size_t)e * 9;
        const float* cp = Cg + (size_t)e * 9;
        float c9[9];
        #pragma unroll
        for (int i = 0; i < 9; ++i) { Fm[i] = fp[i]; c9[i] = cp[i]; }

        // G = F^T F (row-major 3x3)
        float G[9];
        #pragma unroll
        for (int i = 0; i < 3; ++i)
            #pragma unroll
            for (int j = 0; j < 3; ++j)
                G[i*3+j] = Fm[0+i]*Fm[0+j] + Fm[3+i]*Fm[3+j] + Fm[6+i]*Fm[6+j];

        float det = Fm[0]*(Fm[4]*Fm[8]-Fm[5]*Fm[7])
                  - Fm[1]*(Fm[3]*Fm[8]-Fm[5]*Fm[6])
                  + Fm[2]*(Fm[3]*Fm[7]-Fm[4]*Fm[6]);
        float detg = (det < 0.0f) ? 1e-9f : det;

        // Polar decomposition via Higham iteration: X <- 0.5*(X + cof(X)/det(X))
        float X[9];
        #pragma unroll
        for (int i = 0; i < 9; ++i) X[i] = Fm[i];
        #pragma unroll
        for (int it = 0; it < 8; ++it) {
            float c00 = X[4]*X[8]-X[5]*X[7];
            float c01 = X[5]*X[6]-X[3]*X[8];
            float c02 = X[3]*X[7]-X[4]*X[6];
            float c10 = X[2]*X[7]-X[1]*X[8];
            float c11 = X[0]*X[8]-X[2]*X[6];
            float c12 = X[1]*X[6]-X[0]*X[7];
            float c20 = X[1]*X[5]-X[2]*X[4];
            float c21 = X[2]*X[3]-X[0]*X[5];
            float c22 = X[0]*X[4]-X[1]*X[3];
            float d  = X[0]*c00 + X[1]*c01 + X[2]*c02;
            float ad = (fabsf(d) > 1e-12f) ? d : ((d >= 0.0f) ? 1e-12f : -1e-12f);
            float rd = 0.5f * fast_rcp(ad);
            X[0]=0.5f*X[0]+c00*rd; X[1]=0.5f*X[1]+c01*rd; X[2]=0.5f*X[2]+c02*rd;
            X[3]=0.5f*X[3]+c10*rd; X[4]=0.5f*X[4]+c11*rd; X[5]=0.5f*X[5]+c12*rd;
            X[6]=0.5f*X[6]+c20*rd; X[7]=0.5f*X[7]+c21*rd; X[8]=0.5f*X[8]+c22*rd;
        }
        #pragma unroll
        for (int i = 0; i < 9; ++i) Rm[i] = X[i];

        // singular values = sqrt(eig(G)), analytic symmetric 3x3, descending
        float q  = (G[0] + G[4] + G[8]) * (1.0f/3.0f);
        float p1 = G[1]*G[1] + G[2]*G[2] + G[5]*G[5];
        float p2 = (G[0]-q)*(G[0]-q) + (G[4]-q)*(G[4]-q) + (G[8]-q)*(G[8]-q) + 2.0f*p1;
        float p  = sqrtf(fmaxf(p2 * (1.0f/6.0f), 1e-30f));
        float ip = fast_rcp(p);
        float B00=(G[0]-q)*ip, B01=G[1]*ip, B02=G[2]*ip;
        float B11=(G[4]-q)*ip, B12=G[5]*ip, B22=(G[8]-q)*ip;
        float detB = B00*(B11*B22-B12*B12) - B01*(B01*B22-B12*B02) + B02*(B01*B12-B11*B02);
        float r   = fminf(fmaxf(0.5f*detB, -1.0f), 1.0f);
        float phi = acosf(r) * (1.0f/3.0f);
        float e1 = q + 2.0f*p*__cosf(phi);
        float e3 = q + 2.0f*p*__cosf(phi + 2.0943951023931953f);
        float e2 = 3.0f*q - e1 - e3;
        float s0 = sqrtf(fmaxf(e1, 0.0f));
        float s1 = sqrtf(fmaxf(e2, 0.0f));
        float s2 = sqrtf(fmaxf(e3, 0.0f));

        float F00m = fmaxf(Fm[0], 1e-6f);

        // 57-dim feature row (padded to 64) -> f16 in LDS, buffer 0
        _Float16* fr = &sAct[0][t >> 4][t & 15][0];
        fr[0]=(_Float16)s0; fr[1]=(_Float16)s1; fr[2]=(_Float16)s2;
        #pragma unroll
        for (int i = 0; i < 9; ++i) fr[3+i] = (_Float16)G[i];
        fr[12]=(_Float16)detg; fr[13]=(_Float16)__logf(detg);
        fr[14]=(_Float16)F00m; fr[15]=(_Float16)__logf(F00m);
        #pragma unroll
        for (int i = 0; i < 9; ++i) fr[16+i] = (_Float16)c9[i];
        const float* lr = lat + (size_t)traj_ids[0] * 32;
        #pragma unroll
        for (int j = 0; j < 32; ++j) fr[25+j] = (_Float16)lr[j];
        #pragma unroll
        for (int j = 57; j < 64; ++j) fr[j] = (_Float16)0.0f;
    }
    __syncthreads();

    // ---- MLP: each wave runs WMMA chain on its private 16-row tile ----
    const int wave = t >> 5;
    const int lane = t & 31;
    const int row  = lane & 15;   // A: M row / C,D: N column
    const int hi   = lane >> 4;   // half-wave selector
    const int wOff[5] = {0, 4096, 8192, 12288, 16384};
    const int bOff[5] = {0, 64, 128, 192, 256};
    int bufIn = 0;

    #pragma unroll
    for (int layer = 0; layer < 5; ++layer) {
        const _Float16* W = sW + wOff[layer];
        const int ldb    = (layer == 4) ? 16 : 64;
        const int ntiles = (layer == 4) ? 1  : 4;
        #pragma unroll
        for (int nt = 0; nt < ntiles; ++nt) {
            float bv = sB[bOff[layer] + nt*16 + row];
            v8f acc;
            #pragma unroll
            for (int i = 0; i < 8; ++i) acc[i] = bv;
            #pragma unroll
            for (int ks = 0; ks < 2; ++ks) {
                // A fragment: 16-bit A 16x32 layout (lane=M; hi half picks K groups)
                const _Float16* ap = &sAct[bufIn][wave][row][ks*32 + hi*8];
                v8h alo = *(const v8h*)ap;
                v8h ahi = *(const v8h*)(ap + 16);
                v16h a = cat8(alo, ahi);
                // B fragment: lane%16 = N column; elements over K in same order as A
                const _Float16* bp = W + (ks*32 + hi*8)*ldb + nt*16 + row;
                v16h bb;
                #pragma unroll
                for (int s = 0; s < 8; ++s) { bb[s] = bp[s*ldb]; bb[s+8] = bp[(s+16)*ldb]; }
                acc = __builtin_amdgcn_wmma_f32_16x16x32_f16(
                        false, a, false, bb, (short)0, acc, false, false);
            }
            if (layer < 4) {
                #pragma unroll
                for (int v = 0; v < 8; ++v) {
                    int m = v + 8*hi;   // C/D layout: lanes16-31 hold M+8
                    sAct[bufIn ^ 1][wave][m][nt*16 + row] =
                        (_Float16)gelu_fast(acc[v]);
                }
            } else {
                // final 16x16 f32 result -> LDS (stride 18 to dodge conflicts)
                float* dbase = (float*)&sAct[bufIn ^ 1][wave][0][0];
                #pragma unroll
                for (int v = 0; v < 8; ++v) {
                    int m = v + 8*hi;
                    dbase[m*18 + row] = acc[v];
                }
            }
        }
        bufIn ^= 1;
    }
    __syncthreads();

    // ---- epilogue: symmetrize, P = R*x, cauchy = P*F^T ----
    if (t < ROWS_PER_BLOCK && eIdx < nElem) {
        const float* dbase = (const float*)&sAct[1][t >> 4][0][0];
        const float* drow  = dbase + (t & 15) * 18;
        float x[9];
        #pragma unroll
        for (int i = 0; i < 9; ++i) x[i] = drow[i];
        float xs[9];
        #pragma unroll
        for (int i = 0; i < 3; ++i)
            #pragma unroll
            for (int j = 0; j < 3; ++j)
                xs[i*3+j] = 0.5f * (x[i*3+j] + x[j*3+i]);
        float P[9];
        #pragma unroll
        for (int i = 0; i < 3; ++i)
            #pragma unroll
            for (int j = 0; j < 3; ++j)
                P[i*3+j] = Rm[i*3+0]*xs[0*3+j] + Rm[i*3+1]*xs[1*3+j] + Rm[i*3+2]*xs[2*3+j];
        float* op = outg + (size_t)eIdx * 9;
        #pragma unroll
        for (int i = 0; i < 3; ++i)
            #pragma unroll
            for (int j = 0; j < 3; ++j)
                op[i*3+j] = P[i*3+0]*Fm[j*3+0] + P[i*3+1]*Fm[j*3+1] + P[i*3+2]*Fm[j*3+2];
    }
}

extern "C" void kernel_launch(void* const* d_in, const int* in_sizes, int n_in,
                              void* d_out, int out_size, void* d_ws, size_t ws_size,
                              hipStream_t stream) {
    const float* F   = (const float*)d_in[0];
    const float* C   = (const float*)d_in[1];
    const float* w1  = (const float*)d_in[2];
    const float* b1  = (const float*)d_in[3];
    const float* w2  = (const float*)d_in[4];
    const float* b2  = (const float*)d_in[5];
    const float* w3  = (const float*)d_in[6];
    const float* b3  = (const float*)d_in[7];
    const float* w4  = (const float*)d_in[8];
    const float* b4  = (const float*)d_in[9];
    const float* w5  = (const float*)d_in[10];
    const float* b5  = (const float*)d_in[11];
    const float* lat = (const float*)d_in[12];
    const int*   tid = (const int*)d_in[13];
    float* out = (float*)d_out;

    int n = in_sizes[0] / 9;
    int blocks = (n + ROWS_PER_BLOCK - 1) / ROWS_PER_BLOCK;
    hipLaunchKernelGGL(stress_mlp_kernel, dim3(blocks), dim3(NBLK_THREADS), 0, stream,
                       F, C, w1, b1, w2, b2, w3, b3, w4, b4, w5, b5, lat, tid, out, n);
}
